// CausalSelfAttention_1262720385303
// MI455X (gfx1250) — matrix operand order repaired and checked by reading, not compiled, not run on hardware
//
#include <hip/hip_runtime.h>
#include <hip/hip_bf16.h>

// ---- Shapes (compile-time constants from the reference) ----
#define Bsz   2
#define Tlen  2048
#define Cdim  1024
#define NH    16
#define NKV   4
#define HD    64
#define HALF  32
#define QKVN  1536   // 1024 q + 256 k + 256 v

typedef __attribute__((ext_vector_type(16))) _Float16 v16h;
typedef __attribute__((ext_vector_type(8)))  _Float16 v8h;
typedef __attribute__((ext_vector_type(4)))  _Float16 v4h;
typedef __attribute__((ext_vector_type(8)))  float    v8f;
typedef __attribute__((ext_vector_type(4)))  float    v4f;

__device__ __forceinline__ v8f zero8() {
  v8f z = {0.f,0.f,0.f,0.f,0.f,0.f,0.f,0.f};
  return z;
}

__device__ __forceinline__ v8f wmma_f16(v16h a, v16h b, v8f c) {
  // (neg_a, A, neg_b, B, c_mod, C, reuse_a, reuse_b)
  return __builtin_amdgcn_wmma_f32_16x16x32_f16(false, a, false, b, (short)0, c, false, false);
}

// A-fragment (16x32 f16, M = lane%16). k order used here (B uses the same):
//   half h -> K = hi*16 + h   (hi = lane>=16)
// rowbase points at A[row0][k0]; ldk in elements.
__device__ __forceinline__ v16h load_a_frag(const _Float16* rowbase, int ldk, int lane) {
  int m  = lane & 15;
  int hi = (lane >> 4) & 1;
  return *(const v16h*)(rowbase + (size_t)m * ldk + hi * 16);
}

// B-fragment (32x16 f16, N = lane%16) where B[k][n] = W[n][k] (W row-major NxK).
// Same k order as load_a_frag: half h -> K = hi*16 + h  => one contiguous 32B load.
__device__ __forceinline__ v16h load_b_frag(const _Float16* colbase, int ldk, int lane) {
  int n  = lane & 15;
  int hi = (lane >> 4) & 1;
  return *(const v16h*)(colbase + (size_t)n * ldk + hi * 16);
}

// ---------------- Kernel 0: fp32 -> f16 cast (4-wide) ----------------
__global__ void cast_f32_f16(const float* __restrict__ in, _Float16* __restrict__ out, int n4) {
  int i = blockIdx.x * blockDim.x + threadIdx.x;
  if (i < n4) {
    v4f v = *(const v4f*)(in + (size_t)i * 4);
    v4h h = { (_Float16)v.x, (_Float16)v.y, (_Float16)v.z, (_Float16)v.w };
    *(v4h*)(out + (size_t)i * 4) = h;
  }
}

// ---------------- Kernel 1/5: WMMA GEMM  C(MxN) = A(MxK) * W(NxK)^T ----------------
// One wave computes a 64x64 output tile (16 WMMAs / 16 b128 loads per K-step).
__global__ void __launch_bounds__(32)
gemm_f16(const _Float16* __restrict__ A, const _Float16* __restrict__ W,
         float* __restrict__ Cout, int K, int ldOut) {
  const int lane = threadIdx.x;
  const int col0 = blockIdx.x * 64;
  const int row0 = blockIdx.y * 64;

  v8f acc[4][4];
  #pragma unroll
  for (int mi = 0; mi < 4; ++mi)
    #pragma unroll
    for (int ni = 0; ni < 4; ++ni) acc[mi][ni] = zero8();

  for (int k0 = 0; k0 < K; k0 += 32) {
    // software prefetch a few K-steps ahead (global_prefetch_b8; speculative, OOB-safe)
    {
      const _Float16* pa = A + (size_t)(row0 + (lane & 15)) * K + k0 + 128;
      const _Float16* pb = W + (size_t)(col0 + (lane & 15)) * K + k0 + 128;
      __builtin_prefetch(pa, 0, 3);
      __builtin_prefetch(pb, 0, 3);
    }
    v16h af[4], bf[4];
    #pragma unroll
    for (int mi = 0; mi < 4; ++mi)
      af[mi] = load_a_frag(A + (size_t)(row0 + mi * 16) * K + k0, K, lane);
    #pragma unroll
    for (int ni = 0; ni < 4; ++ni)
      bf[ni] = load_b_frag(W + (size_t)(col0 + ni * 16) * K + k0, K, lane);
    #pragma unroll
    for (int mi = 0; mi < 4; ++mi)
      #pragma unroll
      for (int ni = 0; ni < 4; ++ni)
        acc[mi][ni] = wmma_f16(af[mi], bf[ni], acc[mi][ni]);
  }

  // C layout: lanes 0-15 rows r, lanes 16-31 rows r+8; col = lane%16.
  const int n  = lane & 15;
  const int hi = lane >> 4;
  #pragma unroll
  for (int mi = 0; mi < 4; ++mi) {
    #pragma unroll
    for (int ni = 0; ni < 4; ++ni) {
      #pragma unroll
      for (int r = 0; r < 8; ++r) {
        int row = row0 + mi * 16 + hi * 8 + r;
        Cout[(size_t)row * ldOut + col0 + ni * 16 + n] = acc[mi][ni][r];
      }
    }
  }
}

// ---------------- Kernel 2: RoPE + RMSNorm, emit f16 Q/K, transposed f16 V ----------------
// grid = (B*T, NH + 2*NKV), block = 64 (one head)
__global__ void __launch_bounds__(64)
rope_rms(const float* __restrict__ qkv, const float* __restrict__ cosb,
         const float* __restrict__ sinb,
         _Float16* __restrict__ q16, _Float16* __restrict__ k16, _Float16* __restrict__ v16T) {
  const int token = blockIdx.x;          // b*T + t
  const int h     = blockIdx.y;          // 0..15 q, 16..19 k, 20..23 v
  const int d     = threadIdx.x;         // 0..63
  const int b     = token >> 11;
  const int t     = token & (Tlen - 1);

  __shared__ float sv[HD];

  if (h >= NH + NKV) {                   // V: plain cast, store transposed (b,kv,d,t)
    int kv = h - (NH + NKV);
    float val = qkv[(size_t)token * QKVN + 1280 + kv * HD + d];
    v16T[(((size_t)(b * NKV + kv) * HD + d) * Tlen) + t] = (_Float16)val;
    return;
  }

  const float* base = qkv + (size_t)token * QKVN + (h < NH ? h * HD : Cdim + (h - NH) * HD);
  float x = base[d];
  sv[d] = x;
  __syncthreads();

  int dh = d & (HALF - 1);
  float c = cosb[t * HALF + dh];
  float s = sinb[t * HALF + dh];
  float r;
  if (d < HALF) r =  x * c + sv[d + HALF] * s;
  else          r = -sv[d - HALF] * s + x * c;
  __syncthreads();
  sv[d] = r * r;
  __syncthreads();
  float ss = 0.f;
  #pragma unroll
  for (int i = 0; i < HD; ++i) ss += sv[i];
  float o = r * rsqrtf(ss * (1.0f / HD) + 1.1920929e-7f);

  if (h < NH) {
    q16[(((size_t)(b * NH + h) * Tlen + t) * HD) + d] = (_Float16)o;
  } else {
    int kv = h - NH;
    k16[(((size_t)(b * NKV + kv) * Tlen + t) * HD) + d] = (_Float16)o;
  }
}

// ---------------- Kernel 3: causal flash attention (one wave = 16 q rows) ----------------
// grid = (T/16, NH, B), block = 32
__global__ void __launch_bounds__(32)
flash_attn(const _Float16* __restrict__ q16, const _Float16* __restrict__ k16,
           const _Float16* __restrict__ v16T, _Float16* __restrict__ y16) {
  const int lane  = threadIdx.x;
  const int qtile = blockIdx.x;
  const int h     = blockIdx.y;
  const int b     = blockIdx.z;
  const int kv    = h >> 2;              // group size g = NH/NKV = 4
  const int qrow0 = qtile * 16;
  const int n     = lane & 15;
  const int hi    = lane >> 4;

  const _Float16* qbase = q16  + ((size_t)(b * NH  + h ) * Tlen + qrow0) * HD;
  const _Float16* kbase = k16  + ((size_t)(b * NKV + kv) * Tlen) * HD;
  const _Float16* vbase = v16T + ((size_t)(b * NKV + kv) * HD) * Tlen;

  v16h qa0 = load_a_frag(qbase +  0, HD, lane);   // K-dims 0..31
  v16h qa1 = load_a_frag(qbase + 32, HD, lane);   // K-dims 32..63

  float m[8], l[8];
  v8f o0 = zero8(), o1 = zero8(), o2 = zero8(), o3 = zero8();
  #pragma unroll
  for (int r = 0; r < 8; ++r) { m[r] = -1e30f; l[r] = 0.f; }

  __shared__ _Float16 pl[16 * 32];

  const int kend   = qrow0 + 16;
  const int ntiles = (kend + 31) >> 5;
  const float scale = 0.125f;            // 1/sqrt(64)

  for (int tIdx = 0; tIdx < ntiles; ++tIdx) {
    const int j0 = tIdx * 32;

    // ---- scores: S(16x32) = Q(16x64) * K^T ----
    v8f s0 = zero8(), s1 = zero8();
    {
      v16h bk00 = load_b_frag(kbase + (size_t)j0 * HD +  0, HD, lane);
      v16h bk01 = load_b_frag(kbase + (size_t)j0 * HD + 32, HD, lane);
      s0 = wmma_f16(qa0, bk00, s0);
      s0 = wmma_f16(qa1, bk01, s0);
      v16h bk10 = load_b_frag(kbase + (size_t)(j0 + 16) * HD +  0, HD, lane);
      v16h bk11 = load_b_frag(kbase + (size_t)(j0 + 16) * HD + 32, HD, lane);
      s1 = wmma_f16(qa0, bk10, s1);
      s1 = wmma_f16(qa1, bk11, s1);
    }

    // ---- causal mask + online softmax (rows owned by this half-wave) ----
    #pragma unroll
    for (int r = 0; r < 8; ++r) {
      int row = qrow0 + hi * 8 + r;
      int c0  = j0 + n;
      float v0 = (c0      <= row) ? s0[r] * scale : -1e30f;
      float v1 = (c0 + 16 <= row) ? s1[r] * scale : -1e30f;
      float tm = fmaxf(v0, v1);
      tm = fmaxf(tm, __shfl_xor(tm, 1));
      tm = fmaxf(tm, __shfl_xor(tm, 2));
      tm = fmaxf(tm, __shfl_xor(tm, 4));
      tm = fmaxf(tm, __shfl_xor(tm, 8));
      float nm   = fmaxf(m[r], tm);
      float corr = __expf(m[r] - nm);
      float p0   = __expf(v0 - nm);
      float p1   = __expf(v1 - nm);
      float rs = p0 + p1;
      rs += __shfl_xor(rs, 1);
      rs += __shfl_xor(rs, 2);
      rs += __shfl_xor(rs, 4);
      rs += __shfl_xor(rs, 8);
      l[r] = l[r] * corr + rs;
      m[r] = nm;
      o0[r] *= corr; o1[r] *= corr; o2[r] *= corr; o3[r] *= corr;
      // stage P (C-layout -> LDS row-major 16x32) for A-fragment reload
      pl[(hi * 8 + r) * 32 + n]      = (_Float16)p0;
      pl[(hi * 8 + r) * 32 + 16 + n] = (_Float16)p1;
    }
    __syncthreads();

    // ---- O += P(16x32) * V(32x64) ----
    v16h pa = load_a_frag(pl, 32, lane);
    {
      v16h vb0 = load_b_frag(vbase + (size_t)( 0) * Tlen + j0, Tlen, lane);
      v16h vb1 = load_b_frag(vbase + (size_t)(16) * Tlen + j0, Tlen, lane);
      v16h vb2 = load_b_frag(vbase + (size_t)(32) * Tlen + j0, Tlen, lane);
      v16h vb3 = load_b_frag(vbase + (size_t)(48) * Tlen + j0, Tlen, lane);
      o0 = wmma_f16(pa, vb0, o0);
      o1 = wmma_f16(pa, vb1, o1);
      o2 = wmma_f16(pa, vb2, o2);
      o3 = wmma_f16(pa, vb3, o3);
    }
    __syncthreads();
  }

  // ---- epilogue: normalize, store y16[token][h*64 + d] ----
  const size_t outbase = ((size_t)(b * Tlen + qrow0)) * Cdim + h * HD;
  #pragma unroll
  for (int r = 0; r < 8; ++r) {
    size_t rowoff = (size_t)(hi * 8 + r) * Cdim;
    float inv = 1.0f / l[r];
    y16[outbase + rowoff +  0 + n] = (_Float16)(o0[r] * inv);
    y16[outbase + rowoff + 16 + n] = (_Float16)(o1[r] * inv);
    y16[outbase + rowoff + 32 + n] = (_Float16)(o2[r] * inv);
    y16[outbase + rowoff + 48 + n] = (_Float16)(o3[r] * inv);
  }
}

// ---------------- Launch ----------------
extern "C" void kernel_launch(void* const* d_in, const int* in_sizes, int n_in,
                              void* d_out, int out_size, void* d_ws, size_t ws_size,
                              hipStream_t stream) {
  const float* x    = (const float*)d_in[0];
  const float* cosb = (const float*)d_in[1];
  const float* sinb = (const float*)d_in[2];
  const float* Wq   = (const float*)d_in[3];
  const float* Wk   = (const float*)d_in[4];
  const float* Wv   = (const float*)d_in[5];
  const float* Wo   = (const float*)d_in[6];

  char* ws = (char*)d_ws;
  const size_t MB = 1024 * 1024;
  _Float16* x16  = (_Float16*)(ws + 0);          //  8 MB (4096x1024)
  _Float16* wq16 = (_Float16*)(ws +  8 * MB);    //  2 MB
  _Float16* wk16 = (_Float16*)(ws + 10 * MB);    // .5 MB
  _Float16* wv16 = (_Float16*)(ws + 10 * MB + 512 * 1024);
  _Float16* wo16 = (_Float16*)(ws + 11 * MB);    //  2 MB
  float*    qkvf = (float*)   (ws + 16 * MB);    // 24 MB (4096x1536 f32)
  _Float16* q16  = (_Float16*)(ws + 40 * MB);    //  8 MB (b,h,t,d)
  _Float16* k16  = (_Float16*)(ws + 48 * MB);    //  2 MB (b,kv,t,d)
  _Float16* v16T = (_Float16*)(ws + 50 * MB);    //  2 MB (b,kv,d,t)
  _Float16* y16  = (_Float16*)(ws + 52 * MB);    //  8 MB (token, C)

  const int nx = Bsz * Tlen * Cdim;              // 4194304
  cast_f32_f16<<<(nx / 4 + 255) / 256, 256, 0, stream>>>(x, x16, nx / 4);
  cast_f32_f16<<<(Cdim * Cdim / 4 + 255) / 256, 256, 0, stream>>>(Wq, wq16, Cdim * Cdim / 4);
  cast_f32_f16<<<(NKV * HD * Cdim / 4 + 255) / 256, 256, 0, stream>>>(Wk, wk16, NKV * HD * Cdim / 4);
  cast_f32_f16<<<(NKV * HD * Cdim / 4 + 255) / 256, 256, 0, stream>>>(Wv, wv16, NKV * HD * Cdim / 4);
  cast_f32_f16<<<(Cdim * Cdim / 4 + 255) / 256, 256, 0, stream>>>(Wo, wo16, Cdim * Cdim / 4);

  const int Mrows = Bsz * Tlen;                  // 4096
  // QKV projections into fused (4096 x 1536) f32 buffer
  gemm_f16<<<dim3(Cdim / 64,        Mrows / 64), 32, 0, stream>>>(x16, wq16, qkvf,          Cdim, QKVN);
  gemm_f16<<<dim3((NKV * HD) / 64,  Mrows / 64), 32, 0, stream>>>(x16, wk16, qkvf + Cdim,   Cdim, QKVN);
  gemm_f16<<<dim3((NKV * HD) / 64,  Mrows / 64), 32, 0, stream>>>(x16, wv16, qkvf + 1280,   Cdim, QKVN);

  rope_rms<<<dim3(Mrows, NH + 2 * NKV), 64, 0, stream>>>(qkvf, cosb, sinb, q16, k16, v16T);

  flash_attn<<<dim3(Tlen / 16, NH, Bsz), 32, 0, stream>>>(q16, k16, v16T, y16);

  gemm_f16<<<dim3(Cdim / 64, Mrows / 64), 32, 0, stream>>>(y16, wo16, (float*)d_out, Cdim, Cdim);
}
